// Normalgraph_IB_75557064671961
// MI455X (gfx1250) — matrix operand order repaired and checked by reading, not compiled, not run on hardware
//
#include <hip/hip_runtime.h>
#include <hip/hip_bf16.h>
#include <stdint.h>

typedef __attribute__((ext_vector_type(16))) _Float16 v16h;
typedef __attribute__((ext_vector_type(8)))  float    v8f;

#define N_NODES 50000
#define EDIM    64
#define NEDGE   800000
#define WKLEN   9

union FragAB { uint32_t u[8]; v16h h; };

__device__ __forceinline__ uint32_t pack2h(float a, float b) {
  union { _Float16 h[2]; uint32_t u; } t;
  t.h[0] = (_Float16)a; t.h[1] = (_Float16)b;
  return t.u;
}
__device__ __forceinline__ float sigmoidf(float z) { return 1.0f / (1.0f + expf(-z)); }

// ---------------------------------------------------------------------------
// Edge MLP: mask = sigmoid((noise + (relu(cat(x[row],x[col]) @ W1 + b1) @ W2 + b2)) / 0.5)
// One wave per 16-edge tile. A = 16x128 f16 (gathered), B = W1 128x64 f16 in LDS.
// 4 K-chunks x 4 N-tiles of v_wmma_f32_16x16x32_f16, fp32 accum, fused epilogue.
// ---------------------------------------------------------------------------
__global__ __launch_bounds__(256)
void k_tmask(const float* __restrict__ x, const int* __restrict__ row, const int* __restrict__ col,
             const float* __restrict__ W1, const float* __restrict__ b1,
             const float* __restrict__ W2, const float* __restrict__ b2,
             const float* __restrict__ noise, float* __restrict__ maskA,
             float* __restrict__ maskB, int numTiles)
{
  __shared__ uint32_t ldsB[64 * 64];     // [kpair][n] : {W1[2kp][n], W1[2kp+1][n]}
  __shared__ uint32_t ldsA[8 * 16 * 65]; // per-wave: 16 rows x (64 u32 + 1 pad)

  const int tid  = threadIdx.x;
  const int lane = tid & 31;
  const int wid  = tid >> 5;
  const int ln   = lane & 15;
  const bool hi  = lane >= 16;

  for (int s = tid; s < 64 * 64; s += 256) {
    int kp = s >> 6, n = s & 63;
    ldsB[s] = pack2h(W1[(2 * kp) * 64 + n], W1[(2 * kp + 1) * 64 + n]);
  }
  __syncthreads();

  const int tile = blockIdx.x * 8 + wid;
  const bool active = tile < numTiles;
  uint32_t* aw = ldsA + wid * (16 * 65);

  if (active) {
    // two lanes per edge: half 0 -> x[row[e]] (K 0..63), half 1 -> x[col[e]] (K 64..127)
    int m = lane >> 1, half = lane & 1;
    int e = tile * 16 + m;
    int idx = half ? col[e] : row[e];
    const float4* s4 = (const float4*)(x + (long long)idx * 64);
    uint32_t* dst = aw + m * 65 + half * 32;
#pragma unroll
    for (int j = 0; j < 16; ++j) {
      float4 f = s4[j];
      dst[2 * j]     = pack2h(f.x, f.y);
      dst[2 * j + 1] = pack2h(f.z, f.w);
    }
  }
  __syncthreads();
  if (!active) return;

  v8f acc[4];
#pragma unroll
  for (int nt = 0; nt < 4; ++nt)
#pragma unroll
    for (int r = 0; r < 8; ++r) acc[nt][r] = 0.0f;

  const uint32_t* ar = aw + ln * 65;
#pragma unroll
  for (int kc = 0; kc < 4; ++kc) {
    FragAB a;
    uint32_t base = kc * 16 + (hi ? 4u : 0u);
#pragma unroll
    for (int v = 0; v < 4; ++v) a.u[v] = ar[base + v];
#pragma unroll
    for (int v = 0; v < 4; ++v) a.u[4 + v] = ar[base + 8 + v];
    int kp0 = kc * 16 + (hi ? 4 : 0);
#pragma unroll
    for (int nt = 0; nt < 4; ++nt) {
      FragAB bf;
#pragma unroll
      for (int v = 0; v < 8; ++v) bf.u[v] = ldsB[(kp0 + (v & 3) + ((v >> 2) << 3)) * 64 + (nt * 16 + ln)];
      acc[nt] = __builtin_amdgcn_wmma_f32_16x16x32_f16(false, a.h, false, bf.h,
                                                       (short)0, acc[nt], false, false);
    }
  }

  float b1v[4], w2v[4];
#pragma unroll
  for (int nt = 0; nt < 4; ++nt) { b1v[nt] = b1[nt * 16 + ln]; w2v[nt] = W2[nt * 16 + ln]; }
  const float bb2 = b2[0];

  float sv[8];
#pragma unroll
  for (int r = 0; r < 8; ++r) {
    float s = 0.0f;
#pragma unroll
    for (int nt = 0; nt < 4; ++nt) {
      float h = acc[nt][r] + b1v[nt];
      h = h > 0.0f ? h : 0.0f;
      s += h * w2v[nt];
    }
#pragma unroll
    for (int off = 1; off < 16; off <<= 1) s += __shfl_xor(s, off, 32);
    sv[r] = s;
  }

#pragma unroll
  for (int r = 0; r < 8; ++r) {
    if (ln == r) {
      int m = r + (hi ? 8 : 0);
      int e = tile * 16 + m;
      float mval = sigmoidf((noise[e] + sv[r] + bb2) * 2.0f);
      maskA[e] = mval;
      if (maskB) maskB[e] = mval;
    }
  }
}

// ---------------------------------------------------------------------------
// Node MLP: s_mask = sigmoid((s_noise + relu(cur @ W1 + b1) @ W2 + b2) / 0.5)
// K = 64 -> 2 K-chunks x 4 N-tiles WMMA. Also accumulates sum(mask) for s_reg.
// ---------------------------------------------------------------------------
__global__ __launch_bounds__(256)
void k_smask(const float* __restrict__ cur,
             const float* __restrict__ W1, const float* __restrict__ b1,
             const float* __restrict__ W2, const float* __restrict__ b2,
             const float* __restrict__ noise, float* __restrict__ maskOut,
             float* __restrict__ sreg, int numTiles)
{
  __shared__ uint32_t ldsB[32 * 64];
  __shared__ uint32_t ldsA[8 * 16 * 33]; // 16 rows x (32 u32 + 1 pad)

  const int tid  = threadIdx.x;
  const int lane = tid & 31;
  const int wid  = tid >> 5;
  const int ln   = lane & 15;
  const bool hi  = lane >= 16;

  for (int s = tid; s < 32 * 64; s += 256) {
    int kp = s >> 6, n = s & 63;
    ldsB[s] = pack2h(W1[(2 * kp) * 64 + n], W1[(2 * kp + 1) * 64 + n]);
  }
  __syncthreads();

  const int tile = blockIdx.x * 8 + wid;
  const bool active = tile < numTiles;
  uint32_t* aw = ldsA + wid * (16 * 33);

  if (active) {
    int m = lane >> 1, half = lane & 1;
    const float4* s4 = (const float4*)(cur + (long long)(tile * 16 + m) * 64 + half * 32);
    uint32_t* dst = aw + m * 33 + half * 16;
#pragma unroll
    for (int j = 0; j < 8; ++j) {
      float4 f = s4[j];
      dst[2 * j]     = pack2h(f.x, f.y);
      dst[2 * j + 1] = pack2h(f.z, f.w);
    }
  }
  __syncthreads();
  if (!active) return;

  v8f acc[4];
#pragma unroll
  for (int nt = 0; nt < 4; ++nt)
#pragma unroll
    for (int r = 0; r < 8; ++r) acc[nt][r] = 0.0f;

  const uint32_t* ar = aw + ln * 33;
#pragma unroll
  for (int kc = 0; kc < 2; ++kc) {
    FragAB a;
    uint32_t base = kc * 16 + (hi ? 4u : 0u);
#pragma unroll
    for (int v = 0; v < 4; ++v) a.u[v] = ar[base + v];
#pragma unroll
    for (int v = 0; v < 4; ++v) a.u[4 + v] = ar[base + 8 + v];
    int kp0 = kc * 16 + (hi ? 4 : 0);
#pragma unroll
    for (int nt = 0; nt < 4; ++nt) {
      FragAB bf;
#pragma unroll
      for (int v = 0; v < 8; ++v) bf.u[v] = ldsB[(kp0 + (v & 3) + ((v >> 2) << 3)) * 64 + (nt * 16 + ln)];
      acc[nt] = __builtin_amdgcn_wmma_f32_16x16x32_f16(false, a.h, false, bf.h,
                                                       (short)0, acc[nt], false, false);
    }
  }

  float b1v[4], w2v[4];
#pragma unroll
  for (int nt = 0; nt < 4; ++nt) { b1v[nt] = b1[nt * 16 + ln]; w2v[nt] = W2[nt * 16 + ln]; }
  const float bb2 = b2[0];

  float sv[8];
#pragma unroll
  for (int r = 0; r < 8; ++r) {
    float s = 0.0f;
#pragma unroll
    for (int nt = 0; nt < 4; ++nt) {
      float h = acc[nt][r] + b1v[nt];
      h = h > 0.0f ? h : 0.0f;
      s += h * w2v[nt];
    }
#pragma unroll
    for (int off = 1; off < 16; off <<= 1) s += __shfl_xor(s, off, 32);
    sv[r] = s;
  }

#pragma unroll
  for (int r = 0; r < 8; ++r) {
    if (ln == r) {
      int m = r + (hi ? 8 : 0);
      int i = tile * 16 + m;
      float mval = sigmoidf((noise[i] + sv[r] + bb2) * 2.0f);
      maskOut[i] = mval;
      atomicAdd(sreg, mval);
    }
  }
}

// ---------------------------------------------------------------------------
// CSR row-pointer from sorted COO rows: rptr[r] = first e with row[e] >= r,
// rptr[N] = NE. One thread per edge marks its boundary span.
// ---------------------------------------------------------------------------
__global__ void k_rowptr(int* __restrict__ rptr, const int* __restrict__ row,
                         int ne, int nrows) {
  long long stride = (long long)gridDim.x * blockDim.x;
  for (long long t = (long long)blockIdx.x * blockDim.x + threadIdx.x; t <= ne; t += stride) {
    if (t == ne) {
      for (int q = row[ne - 1] + 1; q <= nrows; ++q) rptr[q] = ne;
    } else {
      int r = row[t];
      int prev = (t == 0) ? -1 : row[t - 1];
      for (int q = prev + 1; q <= r; ++q) rptr[q] = (int)t;
    }
  }
}

// ---------------------------------------------------------------------------
// Atomic-free SpMV: one wave per output row, lane covers columns {lane, lane+32}.
// rptr==null -> fixed-degree CSR (random-walk adjacency, degree=WKLEN).
// Fully writes out (zero-degree rows -> 0), so no zero-init pass needed.
// ---------------------------------------------------------------------------
__global__ __launch_bounds__(256)
void k_spmv(float* __restrict__ out, const float* __restrict__ in,
            const int* __restrict__ rptr, const int* __restrict__ col,
            const float* __restrict__ vals, const float* __restrict__ gate,
            int nrows, int fixedDeg)
{
  const int lane = threadIdx.x & 31;
  const int wavesPerBlock = blockDim.x >> 5;
  const int nWaves = gridDim.x * wavesPerBlock;
  for (int r = blockIdx.x * wavesPerBlock + (threadIdx.x >> 5); r < nrows; r += nWaves) {
    int e0, e1;
    if (rptr) { e0 = rptr[r]; e1 = rptr[r + 1]; }
    else      { e0 = r * fixedDeg; e1 = e0 + fixedDeg; }
    float a0 = 0.0f, a1 = 0.0f;
    for (int e = e0; e < e1; ++e) {
      float v = vals[e];
      if (gate) v *= gate[e];
      const float* src = in + (long long)col[e] * 64;
      a0 += v * src[lane];
      a1 += v * src[lane + 32];
    }
    float* dst = out + (long long)r * 64;
    dst[lane]      = a0;
    dst[lane + 32] = a1;
  }
}

__global__ void k_zero(float* __restrict__ p, long long n) {
  long long s = (long long)gridDim.x * blockDim.x;
  for (long long i = (long long)blockIdx.x * blockDim.x + threadIdx.x; i < n; i += s) p[i] = 0.0f;
}
__global__ void k_copy(float* __restrict__ d, const float* __restrict__ sr, long long n) {
  long long s = (long long)gridDim.x * blockDim.x;
  for (long long i = (long long)blockIdx.x * blockDim.x + threadIdx.x; i < n; i += s) d[i] = sr[i];
}
__global__ void k_axpy(float* __restrict__ acc, const float* __restrict__ v, long long n) {
  long long s = (long long)gridDim.x * blockDim.x;
  for (long long i = (long long)blockIdx.x * blockDim.x + threadIdx.x; i < n; i += s) acc[i] += v[i];
}
// out = sm[node]*cur + (1-sm[node])*mp
__global__ void k_mix(float* __restrict__ out, const float* __restrict__ cur,
                      const float* __restrict__ mp, const float* __restrict__ sm, long long n) {
  long long s = (long long)gridDim.x * blockDim.x;
  for (long long i = (long long)blockIdx.x * blockDim.x + threadIdx.x; i < n; i += s) {
    float m = sm[i >> 6];
    out[i] = m * cur[i] + (1.0f - m) * mp[i];
  }
}
// scale both embeddings by 1/(L+1) and emit scalars
__global__ void k_finalize(float* __restrict__ out, const float* __restrict__ sreg, long long n) {
  long long s = (long long)gridDim.x * blockDim.x;
  const float inv3 = 1.0f / 3.0f;
  for (long long i = (long long)blockIdx.x * blockDim.x + threadIdx.x; i < n; i += s) out[i] *= inv3;
  if (blockIdx.x == 0 && threadIdx.x == 0) {
    out[n]     = 0.0f;                                   // t_reg
    out[n + 1] = sreg[0] / ((float)N_NODES * 2.0f);      // s_reg
  }
}

extern "C" void kernel_launch(void* const* d_in, const int* in_sizes, int n_in,
                              void* d_out, int out_size, void* d_ws, size_t ws_size,
                              hipStream_t stream) {
  (void)in_sizes; (void)n_in; (void)out_size; (void)ws_size;
  const float* x      = (const float*)d_in[0];
  const float* vals   = (const float*)d_in[1];
  const float* rwvals = (const float*)d_in[2];
  const float* tW1    = (const float*)d_in[3];
  const float* tb1    = (const float*)d_in[4];
  const float* tW2    = (const float*)d_in[5];
  const float* tb2    = (const float*)d_in[6];
  const float* sW1    = (const float*)d_in[7];
  const float* sb1    = (const float*)d_in[8];
  const float* sW2    = (const float*)d_in[9];
  const float* sb2    = (const float*)d_in[10];
  const float* tnoise = (const float*)d_in[11];
  const float* snoise = (const float*)d_in[12];
  const int* row      = (const int*)d_in[13];
  const int* col      = (const int*)d_in[14];
  const float* dummy_rwrow = (const float*)d_in[15]; (void)dummy_rwrow; // fixed-degree CSR
  const int* rwcol    = (const int*)d_in[16];

  const long long nfeat = (long long)N_NODES * EDIM;   // 3,200,000
  float* out    = (float*)d_out;
  float* embT   = out;
  float* embS   = out + nfeat;
  float* maskTL = out + 2 * nfeat + 2;                 // t_masks[-1] output region

  float* A     = (float*)d_ws;
  float* B     = A + nfeat;
  float* C     = B + nfeat;
  float* tmask = C + nfeat;                            // 2*NEDGE
  float* smask = tmask + 2LL * NEDGE;                  // 2*N
  float* sreg  = smask + 2LL * N_NODES;                // 1
  int*   rptr  = (int*)(sreg + 1);                     // N+1 ints

  const int TB = 256;
  const int gElem = (int)((nfeat + TB - 1) / TB);      // 12500
  const int gSpmv = 2048;                              // 16384 waves, grid-stride rows
  const int gTm   = (NEDGE / 16 + 7) / 8;              // 6250 blocks, 8 waves each
  const int gSm   = (N_NODES / 16 + 7) / 8;            // 391

  k_zero<<<1, 64, 0, stream>>>(sreg, 1);
  k_rowptr<<<(NEDGE / TB) + 1, TB, 0, stream>>>(rptr, row, NEDGE, N_NODES);

  // --- edge masks (depend only on x), both layers; layer 1 also writes d_out ---
  for (int l = 0; l < 2; ++l)
    k_tmask<<<gTm, TB, 0, stream>>>(x, row, col,
        tW1 + (size_t)l * 128 * 64, tb1 + (size_t)l * 64, tW2 + (size_t)l * 64, tb2 + l,
        tnoise + (size_t)l * NEDGE, tmask + (size_t)l * NEDGE,
        (l == 1) ? maskTL : (float*)nullptr, NEDGE / 16);

  // --- mask-learning pass for s-masks: cur = spmm(cur), then node MLP ---
  k_spmv<<<gSpmv, TB, 0, stream>>>(A, x, rptr, col, vals, nullptr, N_NODES, 0);
  k_smask<<<gSm, TB, 0, stream>>>(A, sW1, sb1, sW2, sb2, snoise, smask, sreg, N_NODES / 16);
  k_spmv<<<gSpmv, TB, 0, stream>>>(B, A, rptr, col, vals, nullptr, N_NODES, 0);
  k_smask<<<gSm, TB, 0, stream>>>(B, sW1 + 64 * 64, sb1 + 64, sW2 + 64, sb2 + 1,
                                  snoise + N_NODES, smask + N_NODES, sreg, N_NODES / 16);

  // --- t branch: gated propagation, emb_t accumulated in d_out ---
  k_copy<<<gElem, TB, 0, stream>>>(embT, x, nfeat);
  k_spmv<<<gSpmv, TB, 0, stream>>>(A, x, rptr, col, vals, tmask, N_NODES, 0);
  k_axpy<<<gElem, TB, 0, stream>>>(embT, A, nfeat);
  k_spmv<<<gSpmv, TB, 0, stream>>>(B, A, rptr, col, vals, tmask + NEDGE, N_NODES, 0);
  k_axpy<<<gElem, TB, 0, stream>>>(embT, B, nfeat);

  // --- s branch: gated mix with random-walk mean pooling, then spmm ---
  k_copy<<<gElem, TB, 0, stream>>>(embS, x, nfeat);
  k_spmv<<<gSpmv, TB, 0, stream>>>(C, x, nullptr, rwcol, rwvals, nullptr, N_NODES, WKLEN);
  k_mix<<<gElem, TB, 0, stream>>>(A, x, C, smask, nfeat);
  k_spmv<<<gSpmv, TB, 0, stream>>>(B, A, rptr, col, vals, nullptr, N_NODES, 0);
  k_axpy<<<gElem, TB, 0, stream>>>(embS, B, nfeat);

  k_spmv<<<gSpmv, TB, 0, stream>>>(C, B, nullptr, rwcol, rwvals, nullptr, N_NODES, WKLEN);
  k_mix<<<gElem, TB, 0, stream>>>(A, B, C, smask + N_NODES, nfeat);
  k_spmv<<<gSpmv, TB, 0, stream>>>(B, A, rptr, col, vals, nullptr, N_NODES, 0);
  k_axpy<<<gElem, TB, 0, stream>>>(embS, B, nfeat);

  // --- emb_t/emb_s /= 3, write t_reg and s_reg ---
  k_finalize<<<2 * gElem, TB, 0, stream>>>(out, sreg, 2 * nfeat);
}
